// GraphormerMultiheadAttention_60155311948316
// MI455X (gfx1250) — matrix operand order repaired and checked by reading, not compiled, not run on hardware
//
#include <hip/hip_runtime.h>
#include <stdint.h>

// ---------------------------------------------------------------------------
// Graphormer MHA for MI455X (gfx1250, wave32, WMMA 16x16x32 f16).
// B=2, T=S=2048, E=1024, H=16, D=64.
// Bandwidth-bound on the two f32 bias tensors (~1.07 GB, read once) ->
// flash-attention fusion, NT loads for the streamed biases (attn_mask stays
// cacheable: it is reused by all B*H block-columns and fits in the 192MB L2),
// all matmuls in f16 WMMA with f32 accumulation, K tiles staged into LDS with
// the CDNA5 async-to-LDS path (ASYNCcnt) so the copy bypasses VGPRs.
// ---------------------------------------------------------------------------

typedef __attribute__((ext_vector_type(16))) _Float16 v16h;
typedef __attribute__((ext_vector_type(8)))  float    v8f;

#define E_DIM 1024
#define T_DIM 2048
#define S_DIM 2048
#define H_NUM 16
#define D_DIM 64
#define B_NUM 2
#define M_ROWS (B_NUM * T_DIM)   // 4096

static __device__ __forceinline__ v8f wmma_f16(v16h a, v16h b, v8f c) {
    return __builtin_amdgcn_wmma_f32_16x16x32_f16(false, a, false, b,
                                                  (short)0, c, false, false);
}

// ---------------------------------------------------------------------------
// Kernel 1: q/k/v projection  out = X @ W^T + bias  (f32 in, f16 out,
// re-laid out head-major [b,h,t,d]).  Block = 256 thr = 8 waves; block tile
// 128 rows x 64 cols; W k-slab staged in LDS, shared by all 8 waves.
// ---------------------------------------------------------------------------
__global__ __launch_bounds__(256)
void qkv_proj_wmma(const float* __restrict__ X, const float* __restrict__ W,
                   const float* __restrict__ bias, _Float16* __restrict__ outHM)
{
    const int tid  = threadIdx.x;
    const int lane = tid & 31, wave = tid >> 5;
    const int hi   = lane >> 4, col = lane & 15;
    const int m0   = blockIdx.x * 128 + wave * 16;
    const int n0   = blockIdx.y * 64;

    __shared__ alignas(64) _Float16 lw[64 * 32];   // W tile [n_local][k_local]

    v8f acc[4] = {};

    for (int kk = 0; kk < E_DIM; kk += 32) {
        // stage W[n0..n0+63][kk..kk+31] as f16 (8 contiguous f32 per thread)
        #pragma unroll
        for (int i = 0; i < 8; ++i) {
            const int e  = tid * 8 + i;
            const int nl = e >> 5, kl = e & 31;
            lw[e] = (_Float16)W[(size_t)(n0 + nl) * E_DIM + kk + kl];
        }
        __syncthreads();

        // A fragment: row = m0+col, K contiguous 16hi..16hi+15
        v16h a;
        const float* xrow = X + (size_t)(m0 + col) * E_DIM + kk;
        #pragma unroll
        for (int i = 0; i < 8; ++i) {
            a[i]     = (_Float16)xrow[16 * hi + i];       // K = 16hi .. 16hi+7
            a[8 + i] = (_Float16)xrow[16 * hi + 8 + i];   // K = 16hi+8 ..
        }

        #pragma unroll
        for (int c = 0; c < 4; ++c) {
            v16h bf;
            const _Float16* brow = &lw[(16 * c + col) * 32 + 16 * hi];
            #pragma unroll
            for (int i = 0; i < 16; ++i) bf[i] = brow[i]; // K = 16hi .. +15
            acc[c] = wmma_f16(a, bf, acc[c]);
        }
        __syncthreads();
    }

    // epilogue: bias add, write f16 head-major [b][h][t][d]
    #pragma unroll
    for (int c = 0; c < 4; ++c) {
        const int n  = n0 + 16 * c + col;
        const int hh = n >> 6, d = n & 63;
        const float bn = bias[n];
        #pragma unroll
        for (int r = 0; r < 8; ++r) {
            const int m  = m0 + r + 8 * hi;
            const int bb = m >> 11;          // T = 2048
            const int t  = m & (T_DIM - 1);
            outHM[(((size_t)bb * H_NUM + hh) * T_DIM + t) * D_DIM + d] =
                (_Float16)(acc[c][r] + bn);
        }
    }
}

// ---------------------------------------------------------------------------
// Kernel 2: fused flash attention.  grid = (T/128, B*H); block = 8 waves,
// each wave owns a 16-row Q strip.  S loop in steps of 32; K tile staged via
// async-to-LDS (no VGPR roundtrip), V tile transposed through VGPRs; biases
// streamed with NT loads; online softmax in f32.
// ---------------------------------------------------------------------------
__global__ __launch_bounds__(256)
void flash_attn_wmma(const _Float16* __restrict__ Qh,
                     const _Float16* __restrict__ Kh,
                     const _Float16* __restrict__ Vh,
                     const float* __restrict__ spat,
                     const float* __restrict__ dirb,
                     const unsigned char* __restrict__ kpm,
                     const float* __restrict__ amask,
                     _Float16* __restrict__ Oh)
{
    const int tid  = threadIdx.x;
    const int lane = tid & 31, wave = tid >> 5;
    const int hi   = lane >> 4, col = lane & 15;
    const int bh   = blockIdx.y;           // b*H + h
    const int b    = bh >> 4, h = bh & 15;
    const int t0   = blockIdx.x * 128 + wave * 16;

    const _Float16* Qp = Qh + (size_t)bh * T_DIM * D_DIM;
    const _Float16* Kp = Kh + (size_t)bh * S_DIM * D_DIM;
    const _Float16* Vp = Vh + (size_t)bh * S_DIM * D_DIM;
    const float*    sp = spat + (size_t)bh * T_DIM * S_DIM;
    const float*    dp = dirb + (size_t)bh * T_DIM * S_DIM;
    const unsigned char* pm = kpm + (size_t)b * S_DIM;

    __shared__ alignas(64) _Float16 smK [32 * 64];     // K tile [s_loc][d]
    __shared__ alignas(64) _Float16 smVt[64 * 32];     // V tile transposed [d][s_loc]
    __shared__ alignas(64) _Float16 smP [8 * 16 * 32]; // per-wave P scratch

    // LDS byte offset of smK (generic addr low 32 bits == LDS offset)
    const unsigned smK_base = (unsigned)(uintptr_t)(&smK[0]);

    // Q A-fragments (K = d split 0..31 / 32..63), loaded once
    v16h aq0, aq1;
    {
        const _Float16* qrow = Qp + (size_t)(t0 + col) * D_DIM;
        #pragma unroll
        for (int i = 0; i < 8; ++i) {
            aq0[i]     = qrow[16 * hi + i];
            aq0[8 + i] = qrow[16 * hi + 8 + i];
            aq1[i]     = qrow[32 + 16 * hi + i];
            aq1[8 + i] = qrow[40 + 16 * hi + i];
        }
    }

    v8f o[4] = {};
    float mrun[8], lrun[8];
    #pragma unroll
    for (int r = 0; r < 8; ++r) { mrun[r] = -3.0e38f; lrun[r] = 0.f; }

    for (int s0 = 0; s0 < S_DIM; s0 += 32) {
        __syncthreads();

        // ---- K tile: CDNA5 async copy straight into LDS (16B per thread,
        //      256 threads == 4KB tile), tracked by ASYNCcnt ----
        {
            const _Float16* gsrc = Kp + (size_t)s0 * D_DIM + tid * 8;
            const unsigned   ldst = smK_base + (unsigned)(tid * 16);
            asm volatile("global_load_async_to_lds_b128 %0, %1, off"
                         :: "v"(ldst), "v"(gsrc)
                         : "memory");
        }
        // ---- V tile: copy with transpose through VGPRs ----
        #pragma unroll
        for (int i = 0; i < 8; ++i) {
            const int e = tid * 8 + i;
            smVt[(e & 63) * 32 + (e >> 6)] = Vp[(size_t)s0 * D_DIM + e];
        }
        asm volatile("s_wait_asynccnt 0x0" ::: "memory");
        __syncthreads();

        // scores: 16x32 tile = two 16x16 C fragments
        v8f sc[2] = {};
        #pragma unroll
        for (int c = 0; c < 2; ++c) {
            v16h bk0, bk1;                      // B col n = s_local, K = d
            const _Float16* krow = &smK[(16 * c + col) * 64];
            #pragma unroll
            for (int i = 0; i < 16; ++i) {
                bk0[i] = krow[16 * hi + i];          // d = 0..31
                bk1[i] = krow[32 + 16 * hi + i];     // d = 32..63
            }
            sc[c] = wmma_f16(aq0, bk0, sc[c]);
            sc[c] = wmma_f16(aq1, bk1, sc[c]);
        }

        // biases: spat/dirb are single-use streams -> NT; amask is reused by
        // all B*H block-columns and L2-resident -> regular cached load.
        float pv[2][8];
        #pragma unroll
        for (int c = 0; c < 2; ++c) {
            const int s = s0 + 16 * c + col;
            const unsigned char dead = pm[s];
            #pragma unroll
            for (int r = 0; r < 8; ++r) {
                const int t = t0 + r + 8 * hi;
                const size_t idx = (size_t)t * S_DIM + s;
                float v = sc[c][r] * 0.125f                       // D^-0.5
                        + amask[idx]
                        + __builtin_nontemporal_load(&sp[idx])
                        + __builtin_nontemporal_load(&dp[idx]);
                pv[c][r] = dead ? -3.0e38f : v;
            }
        }

        // online softmax: 16-lane reductions per row (wave32-safe)
        #pragma unroll
        for (int r = 0; r < 8; ++r) {
            float mx = fmaxf(pv[0][r], pv[1][r]);
            mx = fmaxf(mx, __shfl_xor(mx, 1, 32));
            mx = fmaxf(mx, __shfl_xor(mx, 2, 32));
            mx = fmaxf(mx, __shfl_xor(mx, 4, 32));
            mx = fmaxf(mx, __shfl_xor(mx, 8, 32));
            const float mnew = fmaxf(mrun[r], mx);
            const float corr = __expf(mrun[r] - mnew);
            mrun[r] = mnew;
            o[0][r] *= corr; o[1][r] *= corr; o[2][r] *= corr; o[3][r] *= corr;
            const float p0 = __expf(pv[0][r] - mnew);
            const float p1 = __expf(pv[1][r] - mnew);
            float rs = p0 + p1;
            rs += __shfl_xor(rs, 1, 32);
            rs += __shfl_xor(rs, 2, 32);
            rs += __shfl_xor(rs, 4, 32);
            rs += __shfl_xor(rs, 8, 32);
            lrun[r] = lrun[r] * corr + rs;
            pv[0][r] = p0; pv[1][r] = p1;
        }

        // re-layout P: C-fragment -> A-fragment via per-wave LDS scratch.
        // Same-wave DS ops are in-order (ISA sec 7.3), no barrier needed.
        _Float16* pw = smP + wave * (16 * 32);
        #pragma unroll
        for (int c = 0; c < 2; ++c)
            #pragma unroll
            for (int r = 0; r < 8; ++r)
                pw[(r + 8 * hi) * 32 + 16 * c + col] = (_Float16)pv[c][r];

        v16h ap;
        const _Float16* prow = pw + col * 32;
        #pragma unroll
        for (int i = 0; i < 8; ++i) {
            ap[i]     = prow[16 * hi + i];
            ap[8 + i] = prow[16 * hi + 8 + i];
        }

        // O += P x V : 4 WMMAs over d-chunks of 16
        #pragma unroll
        for (int j = 0; j < 4; ++j) {
            v16h bv;
            const _Float16* vcol = &smVt[(16 * j + col) * 32 + 16 * hi];
            #pragma unroll
            for (int i = 0; i < 16; ++i) bv[i] = vcol[i];
            o[j] = wmma_f16(ap, bv, o[j]);
        }

        if (s0 + 32 < S_DIM) {   // hint next bias tile toward L2
            __builtin_prefetch(&sp[(size_t)(t0 + col) * S_DIM + s0 + 32], 0, 0);
            __builtin_prefetch(&dp[(size_t)(t0 + col) * S_DIM + s0 + 32], 0, 0);
        }
    }

    // epilogue: normalize, write f16 (B,T,E) for the output projection
    #pragma unroll
    for (int r = 0; r < 8; ++r) {
        const int t = t0 + r + 8 * hi;
        const float inv = 1.0f / lrun[r];
        _Float16* orow = Oh + ((size_t)b * T_DIM + t) * E_DIM + h * D_DIM;
        orow[ 0 + col] = (_Float16)(o[0][r] * inv);
        orow[16 + col] = (_Float16)(o[1][r] * inv);
        orow[32 + col] = (_Float16)(o[2][r] * inv);
        orow[48 + col] = (_Float16)(o[3][r] * inv);
    }
}

// ---------------------------------------------------------------------------
// Kernel 3: output projection  out = O @ Wo^T + bo   (f16 in, f32 out)
// ---------------------------------------------------------------------------
__global__ __launch_bounds__(256)
void out_proj_wmma(const _Float16* __restrict__ Xh, const float* __restrict__ W,
                   const float* __restrict__ bias, float* __restrict__ out)
{
    const int tid  = threadIdx.x;
    const int lane = tid & 31, wave = tid >> 5;
    const int hi   = lane >> 4, col = lane & 15;
    const int m0   = blockIdx.x * 128 + wave * 16;
    const int n0   = blockIdx.y * 64;

    __shared__ alignas(64) _Float16 lw[64 * 32];

    v8f acc[4] = {};

    for (int kk = 0; kk < E_DIM; kk += 32) {
        #pragma unroll
        for (int i = 0; i < 8; ++i) {
            const int e  = tid * 8 + i;
            const int nl = e >> 5, kl = e & 31;
            lw[e] = (_Float16)W[(size_t)(n0 + nl) * E_DIM + kk + kl];
        }
        __syncthreads();

        v16h a;
        const _Float16* xrow = Xh + (size_t)(m0 + col) * E_DIM + kk;
        #pragma unroll
        for (int i = 0; i < 8; ++i) {
            a[i]     = xrow[16 * hi + i];
            a[8 + i] = xrow[16 * hi + 8 + i];
        }

        #pragma unroll
        for (int c = 0; c < 4; ++c) {
            v16h bf;
            const _Float16* brow = &lw[(16 * c + col) * 32 + 16 * hi];
            #pragma unroll
            for (int i = 0; i < 16; ++i) bf[i] = brow[i];
            acc[c] = wmma_f16(a, bf, acc[c]);
        }
        __syncthreads();
    }

    #pragma unroll
    for (int c = 0; c < 4; ++c) {
        const int n = n0 + 16 * c + col;
        const float bn = bias[n];
        #pragma unroll
        for (int r = 0; r < 8; ++r) {
            const int m = m0 + r + 8 * hi;
            out[(size_t)m * E_DIM + n] = acc[c][r] + bn;
        }
    }
}

// ---------------------------------------------------------------------------
extern "C" void kernel_launch(void* const* d_in, const int* in_sizes, int n_in,
                              void* d_out, int out_size, void* d_ws, size_t ws_size,
                              hipStream_t stream)
{
    const float* query = (const float*)d_in[0];
    const float* key   = (const float*)d_in[1];
    const float* value = (const float*)d_in[2];
    const float* spat  = (const float*)d_in[3];
    const float* dirb  = (const float*)d_in[4];
    const unsigned char* kpm = (const unsigned char*)d_in[5];
    const float* amask = (const float*)d_in[6];
    const float* Wq = (const float*)d_in[7];  const float* bq = (const float*)d_in[8];
    const float* Wk = (const float*)d_in[9];  const float* bk = (const float*)d_in[10];
    const float* Wv = (const float*)d_in[11]; const float* bv = (const float*)d_in[12];
    const float* Wo = (const float*)d_in[13]; const float* bo = (const float*)d_in[14];
    float* out = (float*)d_out;

    // workspace: 4 x 8MB f16 buffers (Qh, Kh, Vh head-major; Oh [B,T,E])
    _Float16* Qh = (_Float16*)d_ws;
    _Float16* Kh = (_Float16*)((char*)d_ws +  8ull * 1024 * 1024);
    _Float16* Vh = (_Float16*)((char*)d_ws + 16ull * 1024 * 1024);
    _Float16* Oh = (_Float16*)((char*)d_ws + 24ull * 1024 * 1024);

    const dim3 blk(256);
    const dim3 gProj(M_ROWS / 128, E_DIM / 64);   // (32, 16)
    const dim3 gAttn(T_DIM / 128, B_NUM * H_NUM); // (16, 32)

    qkv_proj_wmma<<<gProj, blk, 0, stream>>>(query, Wq, bq, Qh);
    qkv_proj_wmma<<<gProj, blk, 0, stream>>>(key,   Wk, bk, Kh);
    qkv_proj_wmma<<<gProj, blk, 0, stream>>>(value, Wv, bv, Vh);
    flash_attn_wmma<<<gAttn, blk, 0, stream>>>(Qh, Kh, Vh, spat, dirb, kpm, amask, Oh);
    out_proj_wmma<<<gProj, blk, 0, stream>>>(Oh, Wo, bo, out);
}
